// Qwen3DecoderLayer_34806414967306
// MI455X (gfx1250) — matrix operand order repaired and checked by reading, not compiled, not run on hardware
//
#include <hip/hip_runtime.h>
#include <hip/hip_bf16.h>

// ---------------- types ----------------
typedef __bf16 bf16;
typedef __attribute__((ext_vector_type(8)))  float v8f;
typedef __attribute__((ext_vector_type(16))) bf16  v16bf;
typedef __attribute__((ext_vector_type(8)))  bf16  v8bf;

#define WMMA_BF16(a, b, c) \
  __builtin_amdgcn_wmma_f32_16x16x32_bf16(false, (a), false, (b), (short)0, (c), false, false)

// ---------------- model constants ----------------
constexpr int B_   = 2;
constexpr int S_   = 1024;
constexpr int H_   = 1024;
constexpr int NH_  = 16;
constexpr int HD_  = 64;
constexpr int NKV_ = 4;
constexpr int E_   = 8;
constexpr int I_   = 1024;
constexpr int T_   = B_ * S_;          // 2048 tokens

// ---------------- CDNA5 async global->LDS helpers ----------------
// LDS byte offset for async VDST: generic shared pointers carry the LDS
// offset in addr[31:0] (flat->LDS aperture mapping).
#define LDS_OFF(p) ((unsigned)(unsigned long long)(p))

__device__ __forceinline__ void async_g2l_b128(unsigned lds_byte_off, const void* gaddr) {
  asm volatile("global_load_async_to_lds_b128 %0, %1, off"
               :: "v"(lds_byte_off), "v"(gaddr) : "memory");
}
template <int N>
__device__ __forceinline__ void wait_async() {
  asm volatile("s_wait_asynccnt %0" :: "i"(N) : "memory");
}

// =====================================================================
// Generic bf16 WMMA GEMM: C[M,N](f32) (+)= scale[row] * A[M,K] @ Bt^T
//  A  [M,K] bf16 row-major (optional row gather via a_row_idx)
//  Bt [N,K] bf16 (weights pre-transposed once) -> contiguous staging
//  - optional C-row scatter (c_row_idx), per-row scale (c_row_scale)
//  - optional device-side row count (m_count) for routed experts
//  Tile 128x128 (256 thr = 8 waves), wave = 32x64, K-step 32,
//  double-buffered async global->LDS staging.
// =====================================================================
__global__ __launch_bounds__(256)
void q3_gemm_bf16(const bf16* __restrict__ A, const bf16* __restrict__ Bt,
                  float* __restrict__ C,
                  int M, int N, int K,
                  const int*   __restrict__ a_row_idx,
                  const int*   __restrict__ c_row_idx,
                  const float* __restrict__ c_row_scale,
                  const int*   __restrict__ m_count,
                  int accum)
{
  __shared__ __align__(32) bf16 ldsA[2][128 * 32];  // [buf][m][k]
  __shared__ __align__(32) bf16 ldsB[2][128 * 32];  // [buf][n][k]

  int Meff = M;
  if (m_count) { int c = *m_count; Meff = c < M ? c : M; }

  const int bm = blockIdx.y * 128;
  const int bn = blockIdx.x * 128;
  if (bm >= Meff) return;

  const int tid  = threadIdx.x;
  const int lane = tid & 31;
  const int wave = tid >> 5;
  const int wm   = (wave & 3) * 32;   // wave row base inside tile
  const int wn   = (wave >> 2) * 64;  // wave col base inside tile
  const int lm   = lane & 15;
  const int half = lane >> 4;

  // ---- staging assignment: thread -> (row, 16-elem k-chunk), hoisted ----
  const int sm   = tid >> 1;          // 0..127
  const int sc16 = (tid & 1) * 16;    // 0 or 16
  int gm   = bm + sm;
  int arow = 0;
  if (gm < Meff) arow = a_row_idx ? a_row_idx[gm] : gm;   // clamp OOB rows to 0
  const bf16* aptr = A  + (long)arow * K + sc16;
  const bf16* bptr = Bt + (long)(bn + sm) * K + sc16;
  const unsigned ldsA_off = LDS_OFF(&ldsA[0][sm * 32 + sc16]);
  const unsigned ldsB_off = LDS_OFF(&ldsB[0][sm * 32 + sc16]);

  auto stage = [&](int kt, int buf) {
    const bf16* ap = aptr + kt * 32;
    const bf16* bp = bptr + kt * 32;
    unsigned la = ldsA_off + (unsigned)buf * (128 * 32 * 2);
    unsigned lb = ldsB_off + (unsigned)buf * (128 * 32 * 2);
    async_g2l_b128(la,      ap);
    async_g2l_b128(la + 16, ap + 8);
    async_g2l_b128(lb,      bp);
    async_g2l_b128(lb + 16, bp + 8);
  };

  v8f acc[2][4];
  #pragma unroll
  for (int i = 0; i < 2; ++i)
    #pragma unroll
    for (int j = 0; j < 4; ++j)
      #pragma unroll
      for (int r = 0; r < 8; ++r) acc[i][j][r] = 0.0f;

  const int nk = K / 32;
  stage(0, 0);
  for (int kt = 0; kt < nk; ++kt) {
    const int cur = kt & 1;
    if (kt + 1 < nk) {
      stage(kt + 1, cur ^ 1);   // prefetch next tile into other buffer
      wait_async<4>();          // retire current tile's 4 asyncs only
    } else {
      wait_async<0>();
    }
    __syncthreads();

    // ---- fragments (ISA 7.12.2 layouts) ----
    const bf16* Abuf = ldsA[cur];
    const bf16* Bbuf = ldsB[cur];
    v16bf afr[2], bfr[4];
    #pragma unroll
    for (int i = 0; i < 2; ++i) {
      const bf16* base = Abuf + (wm + i * 16 + lm) * 32;
      v8bf lo = *(const v8bf*)(base + 8 * half);        // K = 8*half + 0..7
      v8bf hi = *(const v8bf*)(base + 16 + 8 * half);   // K = 16+8*half + 0..7
      #pragma unroll
      for (int q = 0; q < 8; ++q) { afr[i][q] = lo[q]; afr[i][q + 8] = hi[q]; }
    }
    #pragma unroll
    for (int j = 0; j < 4; ++j) {
      const bf16* bb = Bbuf + (wn + j * 16 + lm) * 32;
      v8bf lo = *(const v8bf*)(bb + 8 * half);
      v8bf hi = *(const v8bf*)(bb + 16 + 8 * half);
      #pragma unroll
      for (int q = 0; q < 8; ++q) { bfr[j][q] = lo[q]; bfr[j][q + 8] = hi[q]; }
    }

    #pragma unroll
    for (int i = 0; i < 2; ++i)
      #pragma unroll
      for (int j = 0; j < 4; ++j)
        acc[i][j] = WMMA_BF16(afr[i], bfr[j], acc[i][j]);
    __syncthreads();
  }

  // ---- epilogue (C tile: row = r + 8*half, col = lane&15) ----
  #pragma unroll
  for (int i = 0; i < 2; ++i) {
    #pragma unroll
    for (int r = 0; r < 8; ++r) {
      int grow = bm + wm + i * 16 + r + 8 * half;
      if (grow >= Meff) continue;
      int   orow  = c_row_idx   ? c_row_idx[grow]   : grow;
      float scale = c_row_scale ? c_row_scale[grow] : 1.0f;
      #pragma unroll
      for (int j = 0; j < 4; ++j) {
        int col = bn + wn + j * 16 + lm;
        if (col >= N) continue;
        float v = scale * acc[i][j][r];
        long  o = (long)orow * N + col;
        if (accum) C[o] += v; else C[o] = v;
      }
    }
  }
}

// =====================================================================
// Flash attention (causal, GQA). Block = 128 thr (4 waves), each wave
// owns 16 q-rows; k-tiles of 32 columns; WMMA for QK^T and P@V.
// Q [B,NH,S,HD], K/V [B,NKV,S,HD] bf16 -> O [T, NH*HD] bf16.
// =====================================================================
__global__ __launch_bounds__(128)
void q3_flash_attn(const bf16* __restrict__ Q, const bf16* __restrict__ Kt,
                   const bf16* __restrict__ Vt, bf16* __restrict__ O)
{
  constexpr int QT = S_ / 64;
  const int blk  = blockIdx.x;
  const int b    = blk / (NH_ * QT);
  const int h    = (blk / QT) % NH_;
  const int qt   = blk % QT;
  const int hkv  = h / (NH_ / NKV_);
  const int tid  = threadIdx.x;
  const int lane = tid & 31, wave = tid >> 5;
  const int lm   = lane & 15, half = lane >> 4;
  const int qbase = qt * 64 + wave * 16;

  __shared__ __align__(32) bf16 ldsK[32 * 64];     // [kcol][hd]
  __shared__ __align__(32) bf16 ldsV[64 * 32];     // [vcol][krow] (transposed)
  __shared__ __align__(32) bf16 ldsP[4][16 * 32];  // per-wave P relayout slab

  const bf16* qptr = Q  + ((long)b * NH_  + h)   * S_ * HD_;
  const bf16* kptr = Kt + ((long)b * NKV_ + hkv) * S_ * HD_;
  const bf16* vptr = Vt + ((long)b * NKV_ + hkv) * S_ * HD_;

  // Q A-fragments for hd-chunks 0-31 / 32-63
  v16bf qf[2];
  #pragma unroll
  for (int j = 0; j < 2; ++j) {
    const bf16* base = qptr + (long)(qbase + lm) * HD_ + j * 32;
    v8bf lo = *(const v8bf*)(base + 8 * half);
    v8bf hi = *(const v8bf*)(base + 16 + 8 * half);
    #pragma unroll
    for (int q = 0; q < 8; ++q) { qf[j][q] = lo[q]; qf[j][q + 8] = hi[q]; }
  }

  v8f acc[4];
  #pragma unroll
  for (int j = 0; j < 4; ++j)
    #pragma unroll
    for (int r = 0; r < 8; ++r) acc[j][r] = 0.0f;
  float mrow[8], lrow[8];
  #pragma unroll
  for (int r = 0; r < 8; ++r) { mrow[r] = -1e30f; lrow[r] = 0.0f; }

  const float sm_scale = 0.125f;          // 1/sqrt(64)
  const int   kend     = qt * 64 + 64;    // causal bound for this block

  for (int kb = 0; kb < kend; kb += 32) {
    // stage K tile async (contiguous); V tile transposed via registers
    for (int e = tid; e < 32 * 8; e += 128) {
      int kc = e >> 3, dc = (e & 7) * 8;
      async_g2l_b128(LDS_OFF(&ldsK[kc * 64 + dc]),
                     kptr + (long)(kb + kc) * HD_ + dc);
      v8bf vv = *(const v8bf*)(vptr + (long)(kb + kc) * HD_ + dc);
      #pragma unroll
      for (int q = 0; q < 8; ++q) ldsV[(dc + q) * 32 + kc] = vv[q];
    }
    wait_async<0>();
    __syncthreads();

    // scores: 16 q-rows x 32 k-cols as two 16x16 C tiles, hd reduced by 2 WMMAs
    v8f sc[2];
    #pragma unroll
    for (int nt = 0; nt < 2; ++nt) {
      #pragma unroll
      for (int r = 0; r < 8; ++r) sc[nt][r] = 0.0f;
      #pragma unroll
      for (int j = 0; j < 2; ++j) {
        v16bf kf;
        {
          const bf16* kb2 = &ldsK[(nt * 16 + lm) * 64 + j * 32];
          v8bf lo = *(const v8bf*)(kb2 + 8 * half);
          v8bf hi = *(const v8bf*)(kb2 + 16 + 8 * half);
          #pragma unroll
          for (int q = 0; q < 8; ++q) { kf[q] = lo[q]; kf[q + 8] = hi[q]; }
        }
        sc[nt] = WMMA_BF16(qf[j], kf, sc[nt]);
      }
    }

    // online softmax (row stats reduced across 16-lane groups)
    #pragma unroll
    for (int r = 0; r < 8; ++r) {
      int qrow = qbase + r + 8 * half;
      float mx = -1e30f;
      #pragma unroll
      for (int nt = 0; nt < 2; ++nt) {
        int kcol = kb + nt * 16 + lm;
        float s = sc[nt][r] * sm_scale + (kcol <= qrow ? 0.0f : -1e30f);
        sc[nt][r] = s;
        mx = fmaxf(mx, s);
      }
      for (int off = 8; off >= 1; off >>= 1) mx = fmaxf(mx, __shfl_xor(mx, off, 16));
      float nm   = fmaxf(mrow[r], mx);
      float corr = __expf(mrow[r] - nm);
      float rs   = 0.0f;
      #pragma unroll
      for (int nt = 0; nt < 2; ++nt) {
        float pv = __expf(sc[nt][r] - nm);
        sc[nt][r] = pv;
        rs += pv;
      }
      for (int off = 8; off >= 1; off >>= 1) rs += __shfl_xor(rs, off, 16);
      lrow[r] = lrow[r] * corr + rs;
      mrow[r] = nm;
      #pragma unroll
      for (int j = 0; j < 4; ++j) acc[j][r] *= corr;
    }

    // relayout P (C-frag) -> A-frag via per-wave LDS slab
    bf16* P = ldsP[wave];
    #pragma unroll
    for (int nt = 0; nt < 2; ++nt)
      #pragma unroll
      for (int r = 0; r < 8; ++r)
        P[(r + 8 * half) * 32 + nt * 16 + lm] = (bf16)sc[nt][r];

    v16bf pf;
    {
      const bf16* base = P + lm * 32;
      v8bf lo = *(const v8bf*)(base + 8 * half);
      v8bf hi = *(const v8bf*)(base + 16 + 8 * half);
      #pragma unroll
      for (int q = 0; q < 8; ++q) { pf[q] = lo[q]; pf[q + 8] = hi[q]; }
    }
    #pragma unroll
    for (int j = 0; j < 4; ++j) {
      v16bf vf;
      {
        const bf16* vb2 = &ldsV[(j * 16 + lm) * 32];
        v8bf lo = *(const v8bf*)(vb2 + 8 * half);
        v8bf hi = *(const v8bf*)(vb2 + 16 + 8 * half);
        #pragma unroll
        for (int q = 0; q < 8; ++q) { vf[q] = lo[q]; vf[q + 8] = hi[q]; }
      }
      acc[j] = WMMA_BF16(pf, vf, acc[j]);
    }
    __syncthreads();
  }

  // normalize & store O[b*S+qrow, h*64+col] as bf16
  #pragma unroll
  for (int r = 0; r < 8; ++r) {
    int   qrow = qbase + r + 8 * half;
    float inv  = 1.0f / lrow[r];
    #pragma unroll
    for (int j = 0; j < 4; ++j) {
      int col = j * 16 + lm;
      O[((long)b * S_ + qrow) * (NH_ * HD_) + h * HD_ + col] = (bf16)(acc[j][r] * inv);
    }
  }
}

// ---------------- elementwise / small kernels ----------------
// weight pack: X [nb][K,N] f32 (row-major) -> Y [nb][N,K] bf16 (transposed)
__global__ __launch_bounds__(256)
void q3_cvt_bf16_t(const float* __restrict__ X, bf16* __restrict__ Y,
                   int K, int N, int nb) {
  long idx = (long)blockIdx.x * blockDim.x + threadIdx.x;
  long per = (long)K * N;
  if (idx >= per * nb) return;
  int  bb = (int)(idx / per);
  long r  = idx % per;
  int  k  = (int)(r / N), n = (int)(r % N);
  Y[(long)bb * per + (long)n * K + k] = (bf16)X[idx];
}

__global__ __launch_bounds__(256)
void q3_zero_u32(unsigned* __restrict__ p, long n) {
  long i = (long)blockIdx.x * blockDim.x + threadIdx.x;
  if (i < n) p[i] = 0u;
}

__global__ __launch_bounds__(256)
void q3_rmsnorm(const float* __restrict__ X, const float* __restrict__ W,
                bf16* __restrict__ Y) {
  __shared__ float red[256];
  int t = blockIdx.x;
  const float* row = X + (long)t * H_;
  float ss = 0.0f;
  for (int i = threadIdx.x; i < H_; i += 256) { float v = row[i]; ss += v * v; }
  red[threadIdx.x] = ss; __syncthreads();
  for (int o = 128; o > 0; o >>= 1) {
    if (threadIdx.x < o) red[threadIdx.x] += red[threadIdx.x + o];
    __syncthreads();
  }
  float inv = rsqrtf(red[0] / H_ + 1e-6f);
  for (int i = threadIdx.x; i < H_; i += 256)
    Y[(long)t * H_ + i] = (bf16)(W[i] * row[i] * inv);
}

// rope for q or k: X [T, nh*64] f32 -> Y [B,nh,S,64] bf16
__global__ __launch_bounds__(256)
void q3_rope(const float* __restrict__ X, bf16* __restrict__ Y,
             const int* __restrict__ pos, int nh) {
  long idx = (long)blockIdx.x * blockDim.x + threadIdx.x;
  long total = (long)T_ * nh * HD_;
  if (idx >= total) return;
  int  d  = idx % HD_;
  int  hh = (idx / HD_) % nh;
  long t  = idx / ((long)HD_ * nh);
  int  b  = (int)(t / S_), s = (int)(t % S_);
  float p  = (float)pos[t];
  int   fi = d < 32 ? d : d - 32;
  float fr = p * __powf(1000000.0f, -2.0f * (float)fi / 64.0f);
  float c = __cosf(fr), sn = __sinf(fr);
  long  base = t * (long)(nh * HD_) + (long)hh * HD_;
  float x  = X[base + d];
  float xo = d < 32 ? -X[base + d + 32] : X[base + d - 32];
  Y[(((long)b * nh + hh) * S_ + s) * HD_ + d] = (bf16)(x * c + xo * sn);
}

// v reorder: [T, NKV*64] f32 -> [B,NKV,S,64] bf16
__global__ __launch_bounds__(256)
void q3_reorder_v(const float* __restrict__ X, bf16* __restrict__ Y) {
  long idx = (long)blockIdx.x * blockDim.x + threadIdx.x;
  long total = (long)T_ * NKV_ * HD_;
  if (idx >= total) return;
  int  d  = idx % HD_;
  int  hh = (idx / HD_) % NKV_;
  long t  = idx / ((long)HD_ * NKV_);
  int  b  = (int)(t / S_), s = (int)(t % S_);
  Y[(((long)b * NKV_ + hh) * S_ + s) * HD_ + d] = (bf16)X[idx];
}

__global__ __launch_bounds__(256)
void q3_add(const float* __restrict__ A, const float* __restrict__ Bv,
            float* __restrict__ C, long n) {
  long i = (long)blockIdx.x * blockDim.x + threadIdx.x;
  if (i < n) C[i] = A[i] + Bv[i];
}

// router: logits + softmax + top2 + per-expert token lists + rprobs
__global__ __launch_bounds__(256)
void q3_router(const bf16* __restrict__ X2, const float* __restrict__ Wr,
               float* __restrict__ rprobs, int* __restrict__ counts,
               int* __restrict__ lists, float* __restrict__ lw) {
  __shared__ float red[E_][256];
  int t = blockIdx.x;
  float a[E_];
  #pragma unroll
  for (int e = 0; e < E_; ++e) a[e] = 0.0f;
  for (int i = threadIdx.x; i < H_; i += 256) {
    float xv = (float)X2[(long)t * H_ + i];
    #pragma unroll
    for (int e = 0; e < E_; ++e) a[e] += xv * Wr[i * E_ + e];
  }
  #pragma unroll
  for (int e = 0; e < E_; ++e) red[e][threadIdx.x] = a[e];
  __syncthreads();
  for (int o = 128; o > 0; o >>= 1) {
    if (threadIdx.x < o)
      #pragma unroll
      for (int e = 0; e < E_; ++e) red[e][threadIdx.x] += red[e][threadIdx.x + o];
    __syncthreads();
  }
  if (threadIdx.x == 0) {
    float lg[E_], pr[E_];
    float mx = -1e30f;
    #pragma unroll
    for (int e = 0; e < E_; ++e) { lg[e] = red[e][0]; mx = fmaxf(mx, lg[e]); }
    float den = 0.0f;
    #pragma unroll
    for (int e = 0; e < E_; ++e) { pr[e] = __expf(lg[e] - mx); den += pr[e]; }
    #pragma unroll
    for (int e = 0; e < E_; ++e) { pr[e] /= den; rprobs[(long)t * E_ + e] = pr[e]; }
    int i0 = 0;
    #pragma unroll
    for (int e = 1; e < E_; ++e) if (pr[e] > pr[i0]) i0 = e;
    int i1 = (i0 == 0) ? 1 : 0;
    #pragma unroll
    for (int e = 0; e < E_; ++e) if (e != i0 && pr[e] > pr[i1]) i1 = e;
    int s0 = atomicAdd(&counts[i0], 1);
    lists[i0 * T_ + s0] = t; lw[i0 * T_ + s0] = pr[i0];
    int s1 = atomicAdd(&counts[i1], 1);
    lists[i1 * T_ + s1] = t; lw[i1 * T_ + s1] = pr[i1];
  }
}

__global__ __launch_bounds__(256)
void q3_silumul(const float* __restrict__ G, const float* __restrict__ U,
                bf16* __restrict__ A, const int* __restrict__ count, int cols) {
  long idx = (long)blockIdx.x * blockDim.x + threadIdx.x;
  long total = (long)T_ * cols;
  if (idx >= total) return;
  int row = (int)(idx / cols);
  int cap = count ? *count : T_;
  if (row >= cap) return;
  float g = G[idx], u = U[idx];
  float s = g / (1.0f + __expf(-g));
  A[idx] = (bf16)(s * u);
}

__global__ __launch_bounds__(256)
void q3_sgate(const bf16* __restrict__ X2, const float* __restrict__ Wg,
              float* __restrict__ sg) {
  __shared__ float red[256];
  int t = blockIdx.x;
  float a = 0.0f;
  for (int i = threadIdx.x; i < H_; i += 256)
    a += (float)X2[(long)t * H_ + i] * Wg[i];
  red[threadIdx.x] = a; __syncthreads();
  for (int o = 128; o > 0; o >>= 1) {
    if (threadIdx.x < o) red[threadIdx.x] += red[threadIdx.x + o];
    __syncthreads();
  }
  if (threadIdx.x == 0) sg[t] = 1.0f / (1.0f + __expf(-red[0]));
}

__global__ __launch_bounds__(256)
void q3_final(const float* __restrict__ resid, const float* __restrict__ moe,
              const float* __restrict__ sh, const float* __restrict__ sg,
              float* __restrict__ out) {
  long i = (long)blockIdx.x * blockDim.x + threadIdx.x;
  if (i >= (long)T_ * H_) return;
  long t = i / H_;
  out[i] = resid[i] + moe[i] + sg[t] * sh[i];
}

__global__ __launch_bounds__(256)
void q3_aux(const float* __restrict__ rprobs, float* __restrict__ out) {
  __shared__ float red[E_][256];
  float a[E_];
  #pragma unroll
  for (int e = 0; e < E_; ++e) a[e] = 0.0f;
  for (int t = threadIdx.x; t < T_; t += 256)
    #pragma unroll
    for (int e = 0; e < E_; ++e) a[e] += rprobs[(long)t * E_ + e];
  #pragma unroll
  for (int e = 0; e < E_; ++e) red[e][threadIdx.x] = a[e];
  __syncthreads();
  for (int o = 128; o > 0; o >>= 1) {
    if (threadIdx.x < o)
      #pragma unroll
      for (int e = 0; e < E_; ++e) red[e][threadIdx.x] += red[e][threadIdx.x + o];
    __syncthreads();
  }
  if (threadIdx.x == 0) {
    float aux = 0.0f;
    #pragma unroll
    for (int e = 0; e < E_; ++e) {
      float m = red[e][0] / T_ - 1.0f / E_;
      aux += m * m;
    }
    out[0] = aux / E_;
  }
}

// =====================================================================
// host launcher
// =====================================================================
extern "C" void kernel_launch(void* const* d_in, const int* in_sizes, int n_in,
                              void* d_out, int out_size, void* d_ws, size_t ws_size,
                              hipStream_t stream) {
  (void)in_sizes; (void)n_in; (void)out_size; (void)ws_size;

  const float* hidden = (const float*)d_in[0];
  /* d_in[1] attention_mask: causal mask is applied analytically */
  const int*   pos    = (const int*)d_in[2];
  const float* ln1    = (const float*)d_in[3];
  const float* ln2    = (const float*)d_in[4];
  const float* wq     = (const float*)d_in[5];
  const float* wk     = (const float*)d_in[6];
  const float* wv     = (const float*)d_in[7];
  const float* wo     = (const float*)d_in[8];
  const float* wr     = (const float*)d_in[9];
  const float* eg     = (const float*)d_in[10];
  const float* eu     = (const float*)d_in[11];
  const float* ed     = (const float*)d_in[12];
  const float* sg_w   = (const float*)d_in[13];
  const float* su_w   = (const float*)d_in[14];
  const float* sd_w   = (const float*)d_in[15];
  const float* seg_w  = (const float*)d_in[16];
  float* out = (float*)d_out;

  // ---- workspace bump allocator ----
  char* p = (char*)d_ws;
  auto alloc = [&](size_t bytes) -> char* {
    char* r = p;
    p += (bytes + 255) & ~(size_t)255;
    return r;
  };
  bf16*  xbf     = (bf16*)alloc((size_t)T_ * H_ * 2);
  bf16*  x2bf    = (bf16*)alloc((size_t)T_ * H_ * 2);
  bf16*  wq_b    = (bf16*)alloc((size_t)H_ * NH_ * HD_ * 2);     // [N][K]
  bf16*  wk_b    = (bf16*)alloc((size_t)H_ * NKV_ * HD_ * 2);
  bf16*  wv_b    = (bf16*)alloc((size_t)H_ * NKV_ * HD_ * 2);
  bf16*  wo_b    = (bf16*)alloc((size_t)NH_ * HD_ * H_ * 2);
  bf16*  eg_b    = (bf16*)alloc((size_t)E_ * H_ * I_ * 2);
  bf16*  eu_b    = (bf16*)alloc((size_t)E_ * H_ * I_ * 2);
  bf16*  ed_b    = (bf16*)alloc((size_t)E_ * I_ * H_ * 2);
  bf16*  sg_b    = (bf16*)alloc((size_t)H_ * I_ * 2);
  bf16*  su_b    = (bf16*)alloc((size_t)H_ * I_ * 2);
  bf16*  sd_b    = (bf16*)alloc((size_t)I_ * H_ * 2);
  float* qf      = (float*)alloc((size_t)T_ * NH_ * HD_ * 4);
  float* kf      = (float*)alloc((size_t)T_ * NKV_ * HD_ * 4);
  float* vf      = (float*)alloc((size_t)T_ * NKV_ * HD_ * 4);
  bf16*  qbf     = (bf16*)alloc((size_t)T_ * NH_ * HD_ * 2);
  bf16*  kbf     = (bf16*)alloc((size_t)T_ * NKV_ * HD_ * 2);
  bf16*  vbf     = (bf16*)alloc((size_t)T_ * NKV_ * HD_ * 2);
  bf16*  attnbf  = (bf16*)alloc((size_t)T_ * NH_ * HD_ * 2);
  float* attnout = (float*)alloc((size_t)T_ * H_ * 4);
  float* resid   = (float*)alloc((size_t)T_ * H_ * 4);
  float* rprobs  = (float*)alloc((size_t)T_ * E_ * 4);
  int*   counts  = (int*)alloc((size_t)E_ * 4);
  int*   lists   = (int*)alloc((size_t)E_ * T_ * 4);
  float* lw      = (float*)alloc((size_t)E_ * T_ * 4);
  float* Gc      = (float*)alloc((size_t)T_ * I_ * 4);
  float* Uc      = (float*)alloc((size_t)T_ * I_ * 4);
  bf16*  actbf   = (bf16*)alloc((size_t)T_ * I_ * 2);
  float* moe     = (float*)alloc((size_t)T_ * H_ * 4);
  float* sharedo = (float*)alloc((size_t)T_ * H_ * 4);
  float* sgate   = (float*)alloc((size_t)T_ * 4);

  // cvtT: [nb](K,N) f32 -> [nb](N,K) bf16
  auto cvtT = [&](const float* src, bf16* dst, int K, int N, int nb) {
    long n = (long)K * N * nb;
    q3_cvt_bf16_t<<<(unsigned)((n + 255) / 256), 256, 0, stream>>>(src, dst, K, N, nb);
  };
  auto gemm = [&](const bf16* A, const bf16* Bt, float* C, int M, int N, int K,
                  const int* aidx, const int* cidx, const float* cscale,
                  const int* cnt, int accum) {
    dim3 g((unsigned)(N / 128), (unsigned)((M + 127) / 128));
    q3_gemm_bf16<<<g, 256, 0, stream>>>(A, Bt, C, M, N, K, aidx, cidx, cscale, cnt, accum);
  };

  // ---- weight pack: convert + transpose (idempotent, deterministic) ----
  cvtT(wq, wq_b, H_, NH_ * HD_, 1);
  cvtT(wk, wk_b, H_, NKV_ * HD_, 1);
  cvtT(wv, wv_b, H_, NKV_ * HD_, 1);
  cvtT(wo, wo_b, NH_ * HD_, H_, 1);
  cvtT(eg, eg_b, H_, I_, E_);
  cvtT(eu, eu_b, H_, I_, E_);
  cvtT(ed, ed_b, I_, H_, E_);
  cvtT(sg_w, sg_b, H_, I_, 1);
  cvtT(su_w, su_b, H_, I_, 1);
  cvtT(sd_w, sd_b, I_, H_, 1);

  // ---- attention ----
  q3_rmsnorm<<<T_, 256, 0, stream>>>(hidden, ln1, xbf);
  gemm(xbf, wq_b, qf, T_, NH_ * HD_, H_, nullptr, nullptr, nullptr, nullptr, 0);
  gemm(xbf, wk_b, kf, T_, NKV_ * HD_, H_, nullptr, nullptr, nullptr, nullptr, 0);
  gemm(xbf, wv_b, vf, T_, NKV_ * HD_, H_, nullptr, nullptr, nullptr, nullptr, 0);
  {
    long nq = (long)T_ * NH_ * HD_, nk = (long)T_ * NKV_ * HD_;
    q3_rope<<<(unsigned)((nq + 255) / 256), 256, 0, stream>>>(qf, qbf, pos, NH_);
    q3_rope<<<(unsigned)((nk + 255) / 256), 256, 0, stream>>>(kf, kbf, pos, NKV_);
    q3_reorder_v<<<(unsigned)((nk + 255) / 256), 256, 0, stream>>>(vf, vbf);
  }
  q3_flash_attn<<<B_ * NH_ * (S_ / 64), 128, 0, stream>>>(qbf, kbf, vbf, attnbf);
  gemm(attnbf, wo_b, attnout, T_, H_, NH_ * HD_, nullptr, nullptr, nullptr, nullptr, 0);
  q3_add<<<(T_ * H_ + 255) / 256, 256, 0, stream>>>(hidden, attnout, resid, (long)T_ * H_);

  // ---- MoE: router + routed experts (gather/scatter GEMMs) ----
  q3_rmsnorm<<<T_, 256, 0, stream>>>(resid, ln2, x2bf);
  q3_zero_u32<<<1, 256, 0, stream>>>((unsigned*)counts, E_);
  q3_zero_u32<<<(T_ * H_ + 255) / 256, 256, 0, stream>>>((unsigned*)moe, (long)T_ * H_);
  q3_router<<<T_, 256, 0, stream>>>(x2bf, wr, rprobs, counts, lists, lw);

  for (int e = 0; e < E_; ++e) {
    const int*   idx = lists + e * T_;
    const float* wgt = lw + e * T_;
    const int*   cnt = counts + e;
    gemm(x2bf, eg_b + (long)e * H_ * I_, Gc, T_, I_, H_, idx, nullptr, nullptr, cnt, 0);
    gemm(x2bf, eu_b + (long)e * H_ * I_, Uc, T_, I_, H_, idx, nullptr, nullptr, cnt, 0);
    q3_silumul<<<(T_ * I_ + 255) / 256, 256, 0, stream>>>(Gc, Uc, actbf, cnt, I_);
    gemm(actbf, ed_b + (long)e * I_ * H_, moe, T_, H_, I_, nullptr, idx, wgt, cnt, 1);
  }

  // ---- shared expert ----
  gemm(x2bf, sg_b, Gc, T_, I_, H_, nullptr, nullptr, nullptr, nullptr, 0);
  gemm(x2bf, su_b, Uc, T_, I_, H_, nullptr, nullptr, nullptr, nullptr, 0);
  q3_silumul<<<(T_ * I_ + 255) / 256, 256, 0, stream>>>(Gc, Uc, actbf, nullptr, I_);
  gemm(actbf, sd_b, sharedo, T_, H_, I_, nullptr, nullptr, nullptr, nullptr, 0);
  q3_sgate<<<T_, 256, 0, stream>>>(x2bf, seg_w, sgate);

  // ---- combine + outputs ----
  q3_final<<<(T_ * H_ + 255) / 256, 256, 0, stream>>>(resid, moe, sharedo, sgate, out);
  q3_aux<<<1, 256, 0, stream>>>(rprobs, out + (long)T_ * H_);
}
